// RNN_25907242730189
// MI455X (gfx1250) — matrix operand order repaired and checked by reading, not compiled, not run on hardware
//
#include <hip/hip_runtime.h>
#include <math.h>

// Problem constants (match reference).
#define BB   256   // batch
#define TT   512   // time steps
#define INN  256   // input / output dim
#define HH   1024  // hidden dim

#define NBLK 32    // persistent workgroups (co-resident on MI455X)
#define NTHR 256   // 8 waves of 32 (wave32)
#define NWAVE (NBLK * (NTHR / 32))

typedef __attribute__((ext_vector_type(16))) __bf16  v16bf;
typedef __attribute__((ext_vector_type(8)))  float   v8f;
typedef __attribute__((ext_vector_type(4)))  unsigned u32x4;

union FragU { v16bf v; u32x4 q[2]; };

// ---------------------------------------------------------------------------
// Fragment-linear layouts: one WMMA fragment == 1KB contiguous == 512 bf16.
// Lane l owns bytes [l*32, l*32+32) of its fragment -> every fragment load is
// a single fully-coalesced 1KB wave transaction (2x global_load_b128 / lane).
//
// A-fragment (16x32, CDNA5 ISA): lane l (l<16) = row l, elements
// 0..7 -> K {0..7}, 8..15 -> K {16..23}; lanes 16..31 shift K by +8.
// B-fragment (32x16): lane l (l<16) = col l, elements 0..7 -> K {0..7}, 8..15 -> K {16..23};
// lanes 16..31 shift K by +8.  frag_id = tile16 * (K/32) + kchunk.
__device__ __forceinline__ v16bf load_frag(const __bf16* base, size_t frag, int lane) {
  const __bf16* p = base + frag * 512 + (size_t)lane * 16;
  FragU u;
  u.q[0] = *(const u32x4*)(p);
  u.q[1] = *(const u32x4*)(p + 8);
  return u.v;
}

__device__ __forceinline__ float fast_tanh(float x) {
#if __has_builtin(__builtin_amdgcn_tanhf)
  return __builtin_amdgcn_tanhf(x);          // v_tanh_f32 (CDNA5 trans op)
#elif __has_builtin(__builtin_amdgcn_tanh_f32)
  return __builtin_amdgcn_tanh_f32(x);
#else
  return tanhf(x);
#endif
}

// Accumulate a 32x32 C macro-tile over one (A,W) pair: 4 v_wmma per K-chunk.
__device__ __forceinline__ void mma_pair(const __bf16* A, const __bf16* W, int K,
                                         int row0, int col0, int lane,
                                         v8f& c00, v8f& c01, v8f& c10, v8f& c11) {
  const int KC = K >> 5;                               // 32-wide K chunks
  const size_t aF0 = (size_t)(row0 >> 4) * KC;
  const size_t aF1 = aF0 + KC;
  const size_t bF0 = (size_t)(col0 >> 4) * KC;
  const size_t bF1 = bF0 + KC;
  for (int kc = 0; kc < KC; ++kc) {
    v16bf a0 = load_frag(A, aF0 + kc, lane);
    v16bf a1 = load_frag(A, aF1 + kc, lane);
    v16bf b0 = load_frag(W, bF0 + kc, lane);
    v16bf b1 = load_frag(W, bF1 + kc, lane);
    c00 = __builtin_amdgcn_wmma_f32_16x16x32_bf16(false, a0, false, b0, (short)0, c00, false, false);
    c01 = __builtin_amdgcn_wmma_f32_16x16x32_bf16(false, a0, false, b1, (short)0, c01, false, false);
    c10 = __builtin_amdgcn_wmma_f32_16x16x32_bf16(false, a1, false, b0, (short)0, c10, false, false);
    c11 = __builtin_amdgcn_wmma_f32_16x16x32_bf16(false, a1, false, b1, (short)0, c11, false, false);
  }
}

// One fused GEMM stage: out = act(A1*W1^T [+ A2*W2^T] + bias1 [+ bias2]).
// bf16 output is written in A-fragment layout (next stage's operand) via a
// per-wave LDS 32x32 transpose (C layout: lane=col; A layout: lane=row).
// Optional f32 output (final y) goes to d_out in the harness layout.
__device__ __forceinline__ void gemm_stage(
    const __bf16* A1, int K1, const __bf16* W1,
    const __bf16* A2, int K2, const __bf16* W2,
    const float* bias1, const float* bias2,
    int N, __bf16* outB, float* outF, int ldF, bool act,
    __bf16* myLds)                                      // 1024 bf16 per wave
{
  const int lane = threadIdx.x & 31;
  const int wave = (blockIdx.x << 3) + (threadIdx.x >> 5);
  const int numMT = (BB >> 5) * (N >> 5);               // 32x32 macro-tiles
  const int lo  = lane & 15;
  const int hi8 = (lane & 16) >> 1;                     // 0 or 8

  for (int mt = wave; mt < numMT; mt += NWAVE) {
    const int row0 = (mt & 7) << 5;                     // 8 row-tiles cover BB
    const int col0 = (mt >> 3) << 5;
    v8f c00 = {}, c01 = {}, c10 = {}, c11 = {};
    mma_pair(A1, W1, K1, row0, col0, lane, c00, c01, c10, c11);
    if (A2) mma_pair(A2, W2, K2, row0, col0, lane, c00, c01, c10, c11);

    const float bA = bias1[col0 + lo]      + (bias2 ? bias2[col0 + lo]      : 0.0f);
    const float bB = bias1[col0 + 16 + lo] + (bias2 ? bias2[col0 + 16 + lo] : 0.0f);

    // C layout -> LDS (row-major 32x32 bf16 tile), plus optional f32 store.
#pragma unroll
    for (int r = 0; r < 8; ++r) {
      float v00 = c00[r] + bA, v01 = c01[r] + bB;
      float v10 = c10[r] + bA, v11 = c11[r] + bB;
      if (act) { v00 = fast_tanh(v00); v01 = fast_tanh(v01);
                 v10 = fast_tanh(v10); v11 = fast_tanh(v11); }
      const int m0l = hi8 + r, m1l = 16 + hi8 + r;      // local rows in tile
      myLds[m0l * 32 + lo]      = (__bf16)v00;
      myLds[m0l * 32 + 16 + lo] = (__bf16)v01;
      myLds[m1l * 32 + lo]      = (__bf16)v10;
      myLds[m1l * 32 + 16 + lo] = (__bf16)v11;
      if (outF) {
        const int m0 = row0 + m0l, m1 = row0 + m1l;
        outF[(size_t)m0 * ldF + col0 + lo]      = v00;
        outF[(size_t)m0 * ldF + col0 + 16 + lo] = v01;
        outF[(size_t)m1 * ldF + col0 + lo]      = v10;
        outF[(size_t)m1 * ldF + col0 + 16 + lo] = v11;
      }
    }

    // LDS -> A-fragment layout in global, fully coalesced (32B per lane).
    // Same-wave LDS ops are kept in order (DScnt), no barrier needed.
    const int KCn = N >> 5;
    const int kcN = col0 >> 5;
#pragma unroll
    for (int mtl = 0; mtl < 2; ++mtl) {
      const __bf16* lp = myLds + (size_t)(mtl * 16 + lo) * 32 + hi8;
      u32x4 q0 = *(const u32x4*)(lp);                   // cols hi8 .. hi8+7
      u32x4 q1 = *(const u32x4*)(lp + 16);              // cols hi8+16 .. +23
      const size_t frag = (size_t)((row0 >> 4) + mtl) * KCn + kcN;
      __bf16* dst = outB + frag * 512 + (size_t)lane * 16;
      *(u32x4*)(dst)     = q0;
      *((u32x4*)dst + 1) = q1;
    }
  }
}

// Device-wide barrier: monotonic counter, target = barrier#*NBLK.
// Counter re-zeroed by init kernel each call -> deterministic graph replays.
__device__ __forceinline__ void grid_barrier(unsigned* cnt, unsigned target) {
  __threadfence();
  __syncthreads();
  if (threadIdx.x == 0) {
    __hip_atomic_fetch_add(cnt, 1u, __ATOMIC_ACQ_REL, __HIP_MEMORY_SCOPE_AGENT);
    while (__hip_atomic_load(cnt, __ATOMIC_ACQUIRE, __HIP_MEMORY_SCOPE_AGENT) < target) {
      __builtin_amdgcn_s_sleep(1);
    }
  }
  __syncthreads();
  __threadfence();
}

// ---------------------------------------------------------------------------
// Init-time swizzles: f32 row-major -> bf16 fragment-linear.
__device__ __forceinline__ void swizzleB(const float* W, __bf16* dst, int N, int K,
                                         size_t tid, size_t str) {
  const int KC = K >> 5;
  for (size_t i = tid; i < (size_t)N * K; i += str) {
    const int e    = (int)(i & 15);
    const int lane = (int)((i >> 4) & 31);
    const size_t frag = i >> 9;
    const int nt = (int)(frag / KC), kc = (int)(frag % KC);
    const int n = nt * 16 + (lane & 15);
    const int k = kc * 32 + ((lane & 16) >> 1) + ((e & 8) << 1) + (e & 7);
    dst[i] = (__bf16)W[(size_t)n * K + k];
  }
}

__device__ __forceinline__ void swizzleA(const float* S, __bf16* dst, int M, int K,
                                         size_t tid, size_t str) {
  const int KC = K >> 5;
  for (size_t i = tid; i < (size_t)M * K; i += str) {
    const int e    = (int)(i & 15);
    const int lane = (int)((i >> 4) & 31);
    const size_t frag = i >> 9;
    const int mt = (int)(frag / KC), kc = (int)(frag % KC);
    const int m = mt * 16 + (lane & 15);
    const int k = kc * 32 + ((lane & 16) >> 1) + ((e & 8) << 1) + (e & 7);
    dst[i] = (__bf16)S[(size_t)m * K + k];
  }
}

__global__ __launch_bounds__(NTHR) void rnn_init(
    const float* wih0f, const float* whh0f, const float* wih1f, const float* whh1f,
    const float* fcwf, const float* y0,
    __bf16* wih0, __bf16* whh0, __bf16* wih1, __bf16* whh1, __bf16* fcw,
    __bf16* h0s, __bf16* h1s, __bf16* ys, float* out, unsigned* cnt)
{
  const size_t tid = (size_t)blockIdx.x * blockDim.x + threadIdx.x;
  const size_t str = (size_t)gridDim.x * blockDim.x;
  if (tid == 0) *cnt = 0u;
  swizzleB(wih0f, wih0, HH, INN, tid, str);
  swizzleB(whh0f, whh0, HH, HH,  tid, str);
  swizzleB(wih1f, wih1, HH, HH,  tid, str);
  swizzleB(whh1f, whh1, HH, HH,  tid, str);
  swizzleB(fcwf,  fcw,  INN, HH, tid, str);
  for (size_t i = tid; i < (size_t)2 * BB * HH; i += str) {
    h0s[i] = (__bf16)0.0f; h1s[i] = (__bf16)0.0f;       // zero state (layout-free)
  }
  swizzleA(y0, ys, BB, INN, tid, str);                   // y buffer 0 = step-0 input
  for (size_t i = tid; i < (size_t)BB * INN; i += str) {
    ys[(size_t)BB * INN + i] = (__bf16)0.0f;             // y buffer 1
    out[(i / INN) * ((size_t)TT * INN) + (i % INN)] = y0[i]; // out[:,0,:] = y0
  }
}

// ---------------------------------------------------------------------------
// Persistent kernel: 511 steps x 3 dependent GEMM stages, device barriers
// between stages. All operands are L2-resident (bf16 weights = 7.3 MB).
__global__ __launch_bounds__(NTHR) void rnn_persistent(
    const float* b_ih0, const float* b_hh0, const float* b_ih1, const float* b_hh1,
    const float* fc_b,
    const __bf16* wih0, const __bf16* whh0, const __bf16* wih1, const __bf16* whh1,
    const __bf16* fcw,
    __bf16* h0s, __bf16* h1s, __bf16* ys, float* out, unsigned* cnt)
{
  __shared__ __bf16 ldsT[8 * 1024];                      // 2KB transpose tile / wave
  __bf16* myLds = ldsT + ((threadIdx.x >> 5) << 10);

  unsigned bar = 0;
  for (int t = 1; t < TT; ++t) {
    const int cur = t & 1, prv = cur ^ 1;
    const size_t hOff = (size_t)BB * HH, yOff = (size_t)BB * INN;

    // h0 = tanh(y @ Wih0^T + h0 @ Whh0^T + b_ih0 + b_hh0)
    gemm_stage(ys + prv * yOff, INN, wih0,
               h0s + prv * hOff, HH, whh0,
               b_ih0, b_hh0, HH, h0s + cur * hOff, nullptr, 0, true, myLds);
    grid_barrier(cnt, ++bar * NBLK);

    // h1 = tanh(h0 @ Wih1^T + h1 @ Whh1^T + b_ih1 + b_hh1)
    gemm_stage(h0s + cur * hOff, HH, wih1,
               h1s + prv * hOff, HH, whh1,
               b_ih1, b_hh1, HH, h1s + cur * hOff, nullptr, 0, true, myLds);
    grid_barrier(cnt, ++bar * NBLK);

    // y = h1 @ fcW^T + fc_b  -> d_out[:, t, :] (f32) + bf16 feedback buffer
    gemm_stage(h1s + cur * hOff, HH, fcw,
               nullptr, 0, nullptr,
               fc_b, nullptr, INN, ys + cur * yOff,
               out + (size_t)t * INN, (size_t)TT * INN, false, myLds);
    grid_barrier(cnt, ++bar * NBLK);
  }
}

// ---------------------------------------------------------------------------
extern "C" void kernel_launch(void* const* d_in, const int* in_sizes, int n_in,
                              void* d_out, int out_size, void* d_ws, size_t ws_size,
                              hipStream_t stream) {
  const float* y0    = (const float*)d_in[0];
  // d_in[1] = t (length only; T is compile-time here)
  const float* Wih0  = (const float*)d_in[2];
  const float* Whh0  = (const float*)d_in[3];
  const float* b_ih0 = (const float*)d_in[4];
  const float* b_hh0 = (const float*)d_in[5];
  const float* Wih1  = (const float*)d_in[6];
  const float* Whh1  = (const float*)d_in[7];
  const float* b_ih1 = (const float*)d_in[8];
  const float* b_hh1 = (const float*)d_in[9];
  const float* fcW   = (const float*)d_in[10];
  const float* fc_b  = (const float*)d_in[11];
  float* out = (float*)d_out;

  // Workspace layout (~9.7 MB; all L2-resident on MI455X's 192 MB L2).
  char* base = (char*)d_ws;
  unsigned* cnt = (unsigned*)base;                       // barrier counter
  size_t off = 256;
  __bf16* wih0 = (__bf16*)(base + off); off += (size_t)HH * INN * 2;
  __bf16* whh0 = (__bf16*)(base + off); off += (size_t)HH * HH  * 2;
  __bf16* wih1 = (__bf16*)(base + off); off += (size_t)HH * HH  * 2;
  __bf16* whh1 = (__bf16*)(base + off); off += (size_t)HH * HH  * 2;
  __bf16* fcw  = (__bf16*)(base + off); off += (size_t)INN * HH * 2;
  __bf16* h0s  = (__bf16*)(base + off); off += (size_t)2 * BB * HH * 2;  // double-buffered
  __bf16* h1s  = (__bf16*)(base + off); off += (size_t)2 * BB * HH * 2;
  __bf16* ys   = (__bf16*)(base + off); off += (size_t)2 * BB * INN * 2;
  (void)ws_size; (void)in_sizes; (void)n_in; (void)out_size;

  rnn_init<<<256, NTHR, 0, stream>>>(Wih0, Whh0, Wih1, Whh1, fcW, y0,
                                     wih0, whh0, wih1, whh1, fcw,
                                     h0s, h1s, ys, out, cnt);

  rnn_persistent<<<NBLK, NTHR, 0, stream>>>(b_ih0, b_hh0, b_ih1, b_hh1, fc_b,
                                            wih0, whh0, wih1, whh1, fcw,
                                            h0s, h1s, ys, out, cnt);
}
